// QKVAttention_50577534878337
// MI455X (gfx1250) — matrix operand order repaired and checked by reading, not compiled, not run on hardware
//
#include <hip/hip_runtime.h>
#include <hip/hip_bf16.h>
#include <math.h>

typedef __attribute__((ext_vector_type(16))) _Float16 v16h;
typedef __attribute__((ext_vector_type(8)))  _Float16 v8h;
typedef __attribute__((ext_vector_type(8)))  float    v8f;

#define NB 16
#define CH 64
#define TT 2048
#define NEG_BIG -1e9f
// exp(S-4) trick: Q pre-scaled by 1/sqrt(C) * log2(e); C-initial = -4*log2(e)
#define QSCALE (0.125f * 1.44269504f)
#define CSHIFT (-4.0f * 1.44269504f)

// ---------------------------------------------------------------------------
// Prep: fuse (+pos), scaling, f16 convert; WMMA-operand layouts:
//   Qh[n][t][c], Kh[n][s][c], Vh[n][c][s]
// ---------------------------------------------------------------------------
__global__ __launch_bounds__(256) void qkv_prep(const float* __restrict__ qkv,
                                                const float* __restrict__ pos,
                                                _Float16* __restrict__ Qh,
                                                _Float16* __restrict__ Kh,
                                                _Float16* __restrict__ Vh) {
  int idx = blockIdx.x * blockDim.x + threadIdx.x;
  if (idx >= NB * CH * TT) return;
  int t = idx % TT;
  int c = (idx / TT) % CH;
  int n = idx / (TT * CH);
  size_t nb   = (size_t)n * 3 * CH * TT;
  size_t pidx = (size_t)n * CH * TT + (size_t)c * TT + t;
  float p = pos[pidx];
  float q = qkv[nb + (size_t)c * TT + t] + p;
  float k = qkv[nb + (size_t)(CH + c) * TT + t] + p;
  float v = qkv[nb + (size_t)(2 * CH + c) * TT + t] + p;
  Qh[(size_t)n * TT * CH + (size_t)t * CH + c] = (_Float16)(q * QSCALE);
  Kh[(size_t)n * TT * CH + (size_t)t * CH + c] = (_Float16)k;
  Vh[pidx] = (_Float16)v;
}

// per-lane async copy of 16B global -> LDS (raw LDS offset = low 32 bits of
// the flat shared-aperture address, per CDNA5 aperture mapping)
__device__ __forceinline__ void async_b128(const void* gptr, void* lptr) {
  unsigned lds_off = (unsigned)(size_t)lptr;
  asm volatile("global_load_async_to_lds_b128 %0, %1, off"
               :: "v"(lds_off), "v"(gptr) : "memory");
}

// ---------------------------------------------------------------------------
// Flash attention (no-max softmax): 8 waves/block, wave owns 16 query rows.
// s-loop in tiles of 64, K/V staged in LDS via async loads.
// ---------------------------------------------------------------------------
__global__ __launch_bounds__(256) void flash_attn(const _Float16* __restrict__ Qh,
                                                  const _Float16* __restrict__ Kh,
                                                  const _Float16* __restrict__ Vh,
                                                  const unsigned char* __restrict__ mask,
                                                  float* __restrict__ out) {
  __shared__ __attribute__((aligned(128))) _Float16 Ks[64][CH];      // [s][c]
  __shared__ __attribute__((aligned(128))) _Float16 Vs[CH][64];      // [c][s]
  __shared__ __attribute__((aligned(128))) _Float16 Pf[8][2][512];   // per-wave A-frags

  const int tid   = threadIdx.x;
  const int wave  = tid >> 5;
  const int lane  = tid & 31;
  const int l16   = lane & 15;
  const int lhalf = lane >> 4;
  const int n  = blockIdx.y;
  const int t0 = blockIdx.x * 128 + wave * 16;

  // ---- Q A-fragments (16x32 f16), resident all kernel ----
  v16h aq[2];
  {
    const _Float16* qrow = Qh + ((size_t)n * TT + (size_t)(t0 + l16)) * CH;
#pragma unroll
    for (int h = 0; h < 2; ++h) {
      int cb = h * 32 + lhalf * 16;
      ((v8h*)&aq[h])[0] = *(const v8h*)(qrow + cb);
      ((v8h*)&aq[h])[1] = *(const v8h*)(qrow + cb + 8);
    }
  }

  v8f o[4], osum;
#pragma unroll
  for (int ct = 0; ct < 4; ++ct) o[ct] = (v8f){};
  osum = (v8f){};

  v8f cinit;                       // softmax shift as WMMA C-initial (free)
#pragma unroll
  for (int i = 0; i < 8; ++i) cinit[i] = CSHIFT;
  v16h bones;                      // ones B-operand: row sums via WMMA
#pragma unroll
  for (int j = 0; j < 16; ++j) bones[j] = (_Float16)1.0f;

  // mask rows for this lane: t = t0 + i + 8*lhalf, col = s0 + st*16 + l16
  const unsigned char* mblk =
      mask + (size_t)n * TT * TT + (size_t)(t0 + 8 * lhalf) * TT + l16;

  // staging coords: 4 threads per 64-half row, 32B each
  const int sr = tid >> 2;          // 0..63
  const int scc = (tid & 3) * 16;   // 0,16,32,48 (halves)
  const _Float16* kbase = Kh + ((size_t)n * TT) * CH + (size_t)sr * CH + scc;
  const _Float16* vbase = Vh + ((size_t)n * CH + sr) * TT + scc;

  for (int s0 = 0; s0 < TT; s0 += 64) {
    __syncthreads();
    // ---- async stage K (64x64, [s][c]) and V (64x64, [c][s]) tiles ----
    async_b128(kbase + (size_t)s0 * CH,     &Ks[sr][scc]);
    async_b128(kbase + (size_t)s0 * CH + 8, &Ks[sr][scc + 8]);
    async_b128(vbase + s0,                  &Vs[sr][scc]);
    async_b128(vbase + s0 + 8,              &Vs[sr][scc + 8]);
    asm volatile("s_wait_asynccnt 0" ::: "memory");
    __syncthreads();

    // ---- S = Q^T K (+shift): 4 s-subtiles, K-dim = 64 (2 chained WMMAs) ----
    v8f sacc[4];
#pragma unroll
    for (int st = 0; st < 4; ++st) {
      v16h bk0 = *(const v16h*)&Ks[st * 16 + l16][lhalf * 16];
      v16h bk1 = *(const v16h*)&Ks[st * 16 + l16][32 + lhalf * 16];
      v8f acc = __builtin_amdgcn_wmma_f32_16x16x32_f16(false, aq[0], false, bk0,
                                                       (short)0, cinit, false, false);
      acc = __builtin_amdgcn_wmma_f32_16x16x32_f16(false, aq[1], false, bk1,
                                                   (short)0, acc, false, false);
      sacc[st] = acc;
    }

    // ---- mask + exp2 (no max: statically safe), stash P in A-frag order ----
    const unsigned char* mptr = mblk + s0;
#pragma unroll
    for (int i = 0; i < 8; ++i) {
      int lp = (i + 8 * lhalf + 16 * ((lane >> 3) & 1)) * 16 + (lane & 7);
#pragma unroll
      for (int st = 0; st < 4; ++st) {
        float a = sacc[st][i];
        if (mptr[(size_t)i * TT + st * 16]) a = NEG_BIG;
        float p = __builtin_amdgcn_exp2f(a);
        Pf[wave][st >> 1][lp + 8 * (st & 1)] = (_Float16)p;
      }
    }

    asm volatile("s_wait_dscnt 0" ::: "memory");

    // ---- O += P V^T, row-sums += P * ones : 2 frags x (4 c-tiles + 1) ----
#pragma unroll
    for (int f = 0; f < 2; ++f) {
      v16h ap = *(const v16h*)&Pf[wave][f][lane * 16];
#pragma unroll
      for (int ct = 0; ct < 4; ++ct) {
        v16h bv;
        ((v8h*)&bv)[0] = *(const v8h*)&Vs[ct * 16 + l16][f * 32 + lhalf * 8];
        ((v8h*)&bv)[1] = *(const v8h*)&Vs[ct * 16 + l16][f * 32 + lhalf * 8 + 16];
        o[ct] = __builtin_amdgcn_wmma_f32_16x16x32_f16(false, ap, false, bv,
                                                       (short)0, o[ct], false, false);
      }
      osum = __builtin_amdgcn_wmma_f32_16x16x32_f16(false, ap, false, bones,
                                                    (short)0, osum, false, false);
    }
  }

  // ---- normalize and scatter to out[n][c][t] (every lane has full row sum) ----
  float inv[8];
#pragma unroll
  for (int i = 0; i < 8; ++i) inv[i] = __builtin_amdgcn_rcpf(osum[i]);

  float* obase = out + (size_t)n * CH * TT;
#pragma unroll
  for (int ct = 0; ct < 4; ++ct) {
    int c = ct * 16 + l16;
#pragma unroll
    for (int i = 0; i < 8; ++i) {
      int t = t0 + i + 8 * lhalf;
      obase[(size_t)c * TT + t] = o[ct][i] * inv[i];
    }
  }
}

extern "C" void kernel_launch(void* const* d_in, const int* in_sizes, int n_in,
                              void* d_out, int out_size, void* d_ws, size_t ws_size,
                              hipStream_t stream) {
  const float* qkv          = (const float*)d_in[0];
  const float* pos          = (const float*)d_in[1];
  const unsigned char* mask = (const unsigned char*)d_in[2];  // jax bool -> 1 byte
  float* out                = (float*)d_out;

  _Float16* Qh = (_Float16*)d_ws;
  _Float16* Kh = Qh + (size_t)NB * TT * CH;
  _Float16* Vh = Kh + (size_t)NB * TT * CH;

  int total = NB * CH * TT;
  qkv_prep<<<(total + 255) / 256, 256, 0, stream>>>(qkv, pos, Qh, Kh, Vh);
  flash_attn<<<dim3(TT / 128, NB), 256, 0, stream>>>(Qh, Kh, Vh, mask, out);
}
